// SSM_cha_6528350290351
// MI455X (gfx1250) — compile-verified, matrix operand order checked by reading, not run on hardware
//
#include <hip/hip_runtime.h>
#include <math.h>
#include <stdint.h>

#define D_MODEL  128
#define OUT_CH   128
#define D_STATE  16
#define D_INNER  256
#define D_CONV   4
#define DT_RANK  8
#define LSEQ     4096
#define NBATCH   2
#define MROWS    (NBATCH * LSEQ)   // 8192
#define XDBL_LD  48                // 40 real cols padded to 48 (3 WMMA n-tiles)
#define KPAD     40                // LDS row stride in halfs (80B, 16B-aligned)

typedef __attribute__((ext_vector_type(16))) _Float16 v16h;
typedef __attribute__((ext_vector_type(8)))  _Float16 v8h;
typedef __attribute__((ext_vector_type(8)))  float    v8f;

__device__ __forceinline__ float silu_f(float x)     { return x * (1.0f / (1.0f + __expf(-x))); }
__device__ __forceinline__ float softplus_f(float x) { return (x > 20.0f) ? x : log1pf(__expf(x)); }

// 16-bit 16x32 operand fragment (ISA 7.12.2): per lane the 16 halfs are two
// contiguous 8-half chunks at K = half*8 and K = 16 + half*8.
__device__ __forceinline__ v16h frag_load(const _Float16* rowbase, int half) {
    v8h lo = *(const v8h*)(rowbase + (half << 3));
    v8h hi = *(const v8h*)(rowbase + 16 + (half << 3));
    return __builtin_shufflevector(lo, hi, 0, 1, 2, 3, 4, 5, 6, 7,
                                   8, 9, 10, 11, 12, 13, 14, 15);
}

// Convert 4 f32 -> 4 f16 and store as one 64-bit LDS write (dst 8B-aligned).
__device__ __forceinline__ void store4h(_Float16* dst, float a, float b, float c, float d) {
    union { _Float16 h[4]; uint2 u; } u;
    u.h[0] = (_Float16)a; u.h[1] = (_Float16)b;
    u.h[2] = (_Float16)c; u.h[3] = (_Float16)d;
    *(uint2*)dst = u.u;
}

// Async copy of 4 bytes/lane from global to LDS (ASYNCcnt-tracked).
__device__ __forceinline__ void async_copy_b32(uint32_t lds_byte_off, const float* g) {
    asm volatile("global_load_async_to_lds_b32 %0, %1, off"
                 :: "v"(lds_byte_off), "v"((uint64_t)(uintptr_t)g) : "memory");
}

// ---------------------------------------------------------------------------
// K1: xz = x @ W_in ; x gathered from NCHW x1. M=8192, K=128, N=512.
// Block: 256 thr (8 waves as 2x4), tile 64(M) x 128(N), K-chunk 32.
// A staged [m][k], B staged transposed [n][k]; fragments = 2x ds_load_b128.
// Staging packs K-quads -> ds_store_b64.
// ---------------------------------------------------------------------------
__global__ __launch_bounds__(256) void k_inproj(const float* __restrict__ x1,
                                                const float* __restrict__ W_in,
                                                float* __restrict__ xz) {
    __shared__ __align__(16) _Float16 As[64][KPAD];
    __shared__ __align__(16) _Float16 Bs[128][KPAD];
    const int t    = threadIdx.x;
    const int lane = t & 31, wid = t >> 5;
    const int half = lane >> 4, l16 = lane & 15;
    const int wm = wid >> 2, wn = wid & 3;          // 2 x 4 wave grid
    const int row0 = blockIdx.x * 64;
    const int col0 = blockIdx.y * 128;

    v8f acc[2][2] = {};
    for (int kk = 0; kk < D_MODEL; kk += 32) {
        // A[r][c] = x1[b, kk+c, l] (transpose gather; threads sweep l -> coalesced)
        #pragma unroll
        for (int i = 0; i < 2; ++i) {               // 64 r x 8 K-quads = 512
            int idx = t + i * 256;
            int r = idx & 63, cq = idx >> 6;
            int gr = row0 + r;
            int b = gr >> 12, l = gr & (LSEQ - 1);
            const float* src = &x1[((size_t)(b * D_MODEL + kk + cq * 4)) * LSEQ + l];
            store4h(&As[r][cq * 4], src[0], src[LSEQ], src[2 * LSEQ], src[3 * LSEQ]);
        }
        // B transposed: Bs[n][c] = W_in[kk+c][col0+n]
        #pragma unroll
        for (int i = 0; i < 4; ++i) {               // 128 n x 8 K-quads = 1024
            int idx = t + i * 256;
            int n = idx & 127, cq = idx >> 7;
            const float* src = &W_in[(size_t)(kk + cq * 4) * (2 * D_INNER) + col0 + n];
            store4h(&Bs[n][cq * 4], src[0], src[2 * D_INNER],
                    src[4 * D_INNER], src[6 * D_INNER]);
        }
        if (kk + 32 < D_MODEL)
            __builtin_prefetch(&W_in[(size_t)(kk + 32 + (t >> 3)) * (2 * D_INNER) +
                                     col0 + ((t & 7) << 4)], 0, 0);
        __syncthreads();

        v16h af[2], bf[2];
        #pragma unroll
        for (int sm = 0; sm < 2; ++sm)
            af[sm] = frag_load(&As[wm * 32 + sm * 16 + l16][0], half);
        #pragma unroll
        for (int sn = 0; sn < 2; ++sn)
            bf[sn] = frag_load(&Bs[wn * 32 + sn * 16 + l16][0], half);

        #pragma unroll
        for (int sm = 0; sm < 2; ++sm)
            #pragma unroll
            for (int sn = 0; sn < 2; ++sn)
                acc[sm][sn] = __builtin_amdgcn_wmma_f32_16x16x32_f16(
                    false, af[sm], false, bf[sn], (short)0, acc[sm][sn], false, false);
        __syncthreads();
    }
    #pragma unroll
    for (int sm = 0; sm < 2; ++sm)
        #pragma unroll
        for (int sn = 0; sn < 2; ++sn)
            #pragma unroll
            for (int vr = 0; vr < 8; ++vr) {
                int m = wm * 32 + sm * 16 + vr + (half << 3);
                int n = wn * 32 + sn * 16 + l16;
                xz[(size_t)(row0 + m) * (2 * D_INNER) + col0 + n] = acc[sm][sn][vr];
            }
}

// ---------------------------------------------------------------------------
// K2: causal depthwise conv (K=4) over L + bias + SiLU.  xin = xz[:, :256]
// ---------------------------------------------------------------------------
__global__ __launch_bounds__(256) void k_conv(const float* __restrict__ xz,
                                              const float* __restrict__ conv_w,
                                              const float* __restrict__ conv_b,
                                              float* __restrict__ xc) {
    int i = blockIdx.x * 256 + threadIdx.x;
    if (i >= MROWS * D_INNER) return;
    int c   = i & (D_INNER - 1);
    int row = i >> 8;
    int b = row >> 12, l = row & (LSEQ - 1);
    float acc = conv_b[c];
    #pragma unroll
    for (int tap = 0; tap < D_CONV; ++tap) {
        int lt = l - (D_CONV - 1) + tap;
        if (lt >= 0)
            acc += xz[(size_t)(b * LSEQ + lt) * (2 * D_INNER) + c] * conv_w[c * D_CONV + tap];
    }
    xc[(size_t)row * D_INNER + c] = silu_f(acc);
}

// ---------------------------------------------------------------------------
// K3: xdbl = xc @ W_x  (M=8192, K=256, N=40 padded to 48).
// Block: 256 thr, BM=128 (wave wid owns 16 rows), 3 N-tiles per wave.
// A staged from contiguous rows via global_load_b128.
// ---------------------------------------------------------------------------
__global__ __launch_bounds__(256) void k_xproj(const float* __restrict__ xc,
                                               const float* __restrict__ W_x,
                                               float* __restrict__ xdbl) {
    __shared__ __align__(16) _Float16 As[128][KPAD];
    __shared__ __align__(16) _Float16 Bs[48][KPAD];
    const int t    = threadIdx.x;
    const int lane = t & 31, wid = t >> 5;
    const int half = lane >> 4, l16 = lane & 15;
    const int row0 = blockIdx.x * 128;
    const int NREAL = DT_RANK + 2 * D_STATE;   // 40

    v8f acc[3] = {};
    for (int kk = 0; kk < D_INNER; kk += 32) {
        #pragma unroll
        for (int i = 0; i < 4; ++i) {               // 128 r x 8 K-quads = 1024
            int idx = t + i * 256;
            int r = idx >> 3, cq = idx & 7;
            float4 g = *(const float4*)&xc[(size_t)(row0 + r) * D_INNER + kk + cq * 4];
            store4h(&As[r][cq * 4], g.x, g.y, g.z, g.w);
        }
        #pragma unroll
        for (int i = 0; i < 2; ++i) {               // 48 n x 8 K-quads = 384
            int idx = t + i * 256;
            if (idx < 384) {
                int n = idx % 48, cq = idx / 48;
                float v0 = 0.f, v1 = 0.f, v2 = 0.f, v3 = 0.f;
                if (n < NREAL) {
                    const float* src = &W_x[(size_t)(kk + cq * 4) * NREAL + n];
                    v0 = src[0]; v1 = src[NREAL]; v2 = src[2 * NREAL]; v3 = src[3 * NREAL];
                }
                store4h(&Bs[n][cq * 4], v0, v1, v2, v3);
            }
        }
        if (kk + 32 < D_INNER)
            __builtin_prefetch(&xc[(size_t)(row0 + (t >> 1)) * D_INNER + kk + 32 +
                                   ((t & 1) << 4)], 0, 0);
        __syncthreads();

        v16h af = frag_load(&As[wid * 16 + l16][0], half);
        v16h bf[3];
        #pragma unroll
        for (int tn = 0; tn < 3; ++tn)
            bf[tn] = frag_load(&Bs[tn * 16 + l16][0], half);
        #pragma unroll
        for (int tn = 0; tn < 3; ++tn)
            acc[tn] = __builtin_amdgcn_wmma_f32_16x16x32_f16(
                false, af, false, bf[tn], (short)0, acc[tn], false, false);
        __syncthreads();
    }
    #pragma unroll
    for (int tn = 0; tn < 3; ++tn)
        #pragma unroll
        for (int vr = 0; vr < 8; ++vr) {
            int m = wid * 16 + vr + (half << 3);
            int n = tn * 16 + l16;
            xdbl[(size_t)(row0 + m) * XDBL_LD + n] = acc[tn][vr];
        }
}

// ---------------------------------------------------------------------------
// K4: dt = softplus(xdbl[:, :8] @ W_dt + b_dt)   (K=8; scalar dot)
// ---------------------------------------------------------------------------
__global__ __launch_bounds__(256) void k_dt(const float* __restrict__ xdbl,
                                            const float* __restrict__ W_dt,
                                            const float* __restrict__ b_dt,
                                            float* __restrict__ dtb) {
    int i = blockIdx.x * 256 + threadIdx.x;
    if (i >= MROWS * D_INNER) return;
    int c   = i & (D_INNER - 1);
    int row = i >> 8;
    float s = b_dt[c];
    #pragma unroll
    for (int r = 0; r < DT_RANK; ++r)
        s += xdbl[(size_t)row * XDBL_LD + r] * W_dt[r * D_INNER + c];
    dtb[(size_t)row * D_INNER + c] = softplus_f(s);
}

// ---------------------------------------------------------------------------
// K5: selective scan. One lane per (channel c, state s); 16-lane shuffle
// reduction for y = sum_s h*C. B/C row data (32 contiguous floats, cols 8..39)
// is double-buffered into per-wave LDS with global_load_async_to_lds_b32
// (lane supplies global addr + LDS dest; s_wait_asynccnt 1 gates consumption).
// Fused: +xc*D, gate with silu(z). 64 blocks x 128 thr.
// ---------------------------------------------------------------------------
__global__ __launch_bounds__(128) void k_scan(const float* __restrict__ xz,
                                              const float* __restrict__ xc,
                                              const float* __restrict__ xdbl,
                                              const float* __restrict__ dtb,
                                              const float* __restrict__ A_log,
                                              const float* __restrict__ Dvec,
                                              float* __restrict__ yg) {
    __shared__ __align__(16) float bc[4][2][32];   // [wave][slot][col]
    const int lane = threadIdx.x & 31;
    const int wv   = threadIdx.x >> 5;
    const int s    = lane & 15;
    const int b    = blockIdx.x >> 5;                 // 2 batches x 32 groups
    const int cg   = blockIdx.x & 31;
    const int c    = cg * 8 + wv * 2 + (lane >> 4);

    const float Ac = -__expf(A_log[c * D_STATE + s]); // A = -exp(A_log)
    const float Dc = Dvec[c];
    float h = 0.0f;
    const size_t rbase = (size_t)b * LSEQ;

    // LDS byte offsets for this lane's two ring slots (generic ptr low 32 bits)
    const uint32_t lds0 = (uint32_t)(uintptr_t)&bc[wv][0][lane];
    const uint32_t lds1 = (uint32_t)(uintptr_t)&bc[wv][1][lane];
    // prologue: prefetch B|C for rows 0 and 1 (cols 8..39 == 8+lane)
    async_copy_b32(lds0, xdbl + (rbase + 0) * XDBL_LD + 8 + lane);
    async_copy_b32(lds1, xdbl + (rbase + 1) * XDBL_LD + 8 + lane);

    for (int l = 0; l < LSEQ; ++l) {
        const size_t row = rbase + l;
        const float dt_v = dtb[row * D_INNER + c];
        const float xc_v = xc[row * D_INNER + c];
        asm volatile("s_wait_asynccnt 1" ::: "memory");   // copy for row l done
        const float Bv = bc[wv][l & 1][s];
        const float Cv = bc[wv][l & 1][16 + s];
        h = __expf(dt_v * Ac) * h + (dt_v * Bv) * xc_v;
        float yp = h * Cv;
        if (l + 2 < LSEQ)   // reuse slot l&1 for row l+2
            async_copy_b32((l & 1) ? lds1 : lds0,
                           xdbl + (row + 2) * XDBL_LD + 8 + lane);
        yp += __shfl_xor(yp, 8, 16);
        yp += __shfl_xor(yp, 4, 16);
        yp += __shfl_xor(yp, 2, 16);
        yp += __shfl_xor(yp, 1, 16);
        if (s == 0) {
            const float zv = xz[row * (2 * D_INNER) + D_INNER + c];
            yg[row * D_INNER + c] = (yp + xc_v * Dc) * silu_f(zv);
        }
    }
}

// ---------------------------------------------------------------------------
// K6: out = LN(yg @ W_out) -> NCHW. Block owns 16 full rows (N=128 in LDS),
// wave wid owns one 16-col N-tile; fused LayerNorm + transposed store.
// ---------------------------------------------------------------------------
__global__ __launch_bounds__(256) void k_out_ln(const float* __restrict__ yg,
                                                const float* __restrict__ W_out,
                                                const float* __restrict__ ln_g,
                                                const float* __restrict__ ln_b,
                                                float* __restrict__ out) {
    __shared__ __align__(16) _Float16 As[16][KPAD];
    __shared__ __align__(16) _Float16 Bs[128][KPAD];
    __shared__ float Os[16][OUT_CH];
    const int t    = threadIdx.x;
    const int lane = t & 31, wid = t >> 5;
    const int half = lane >> 4, l16 = lane & 15;
    const int row0 = blockIdx.x * 16;

    v8f acc = {};
    for (int kk = 0; kk < D_INNER; kk += 32) {
        if (t < 128) {                               // 16 r x 8 K-quads = 128
            int r = t >> 3, cq = t & 7;
            float4 g = *(const float4*)&yg[(size_t)(row0 + r) * D_INNER + kk + cq * 4];
            store4h(&As[r][cq * 4], g.x, g.y, g.z, g.w);
        }
        #pragma unroll
        for (int i = 0; i < 4; ++i) {                // 128 n x 8 K-quads = 1024
            int idx = t + i * 256;
            int n = idx & 127, cq = idx >> 7;
            const float* src = &W_out[(size_t)(kk + cq * 4) * OUT_CH + n];
            store4h(&Bs[n][cq * 4], src[0], src[OUT_CH], src[2 * OUT_CH], src[3 * OUT_CH]);
        }
        if (kk + 32 < D_INNER)
            __builtin_prefetch(&W_out[(size_t)(kk + 32 + (t >> 3)) * OUT_CH +
                                      ((t & 7) << 4)], 0, 0);
        __syncthreads();
        v16h af = frag_load(&As[l16][0], half);
        v16h bf = frag_load(&Bs[wid * 16 + l16][0], half);
        acc = __builtin_amdgcn_wmma_f32_16x16x32_f16(false, af, false, bf, (short)0, acc,
                                                     false, false);
        __syncthreads();
    }
    #pragma unroll
    for (int vr = 0; vr < 8; ++vr)
        Os[vr + (half << 3)][wid * 16 + l16] = acc[vr];
    __syncthreads();

    // LayerNorm over 128 cols, 16 threads per row (rows map to 16-lane groups)
    const int r = t >> 4, sub = t & 15;
    float s1 = 0.0f, s2 = 0.0f;
    #pragma unroll
    for (int j = 0; j < 8; ++j) { float v = Os[r][sub + 16 * j]; s1 += v; s2 += v * v; }
    s1 += __shfl_xor(s1, 8, 16);  s2 += __shfl_xor(s2, 8, 16);
    s1 += __shfl_xor(s1, 4, 16);  s2 += __shfl_xor(s2, 4, 16);
    s1 += __shfl_xor(s1, 2, 16);  s2 += __shfl_xor(s2, 2, 16);
    s1 += __shfl_xor(s1, 1, 16);  s2 += __shfl_xor(s2, 1, 16);
    const float mean = s1 * (1.0f / OUT_CH);
    const float var  = s2 * (1.0f / OUT_CH) - mean * mean;
    const float rstd = rsqrtf(var + 1e-5f);
    const int gr = row0 + r;
    const int b = gr >> 12, l = gr & (LSEQ - 1);
    #pragma unroll
    for (int j = 0; j < 8; ++j) {
        int col = sub + 16 * j;
        float v = (Os[r][col] - mean) * rstd * ln_g[col] + ln_b[col];
        out[((size_t)(b * OUT_CH + col)) * LSEQ + l] = v;   // NCHW
    }
}

// ---------------------------------------------------------------------------
extern "C" void kernel_launch(void* const* d_in, const int* in_sizes, int n_in,
                              void* d_out, int out_size, void* d_ws, size_t ws_size,
                              hipStream_t stream) {
    const float* x1     = (const float*)d_in[0];
    const float* W_in   = (const float*)d_in[1];
    const float* conv_w = (const float*)d_in[2];
    const float* conv_b = (const float*)d_in[3];
    const float* W_x    = (const float*)d_in[4];
    const float* W_dt   = (const float*)d_in[5];
    const float* b_dt   = (const float*)d_in[6];
    const float* A_log  = (const float*)d_in[7];
    const float* Dv     = (const float*)d_in[8];
    const float* W_out  = (const float*)d_in[9];
    const float* ln_g   = (const float*)d_in[10];
    const float* ln_b   = (const float*)d_in[11];
    float* out = (float*)d_out;

    char* ws = (char*)d_ws;
    size_t off = 0;
    float* xz   = (float*)(ws + off); off += (size_t)MROWS * (2 * D_INNER) * sizeof(float);
    float* xc   = (float*)(ws + off); off += (size_t)MROWS * D_INNER * sizeof(float);
    float* xdbl = (float*)(ws + off); off += (size_t)MROWS * XDBL_LD * sizeof(float);
    float* dtb  = (float*)(ws + off); off += (size_t)MROWS * D_INNER * sizeof(float);
    float* yg   = (float*)(ws + off); off += (size_t)MROWS * D_INNER * sizeof(float);
    (void)ws_size; (void)in_sizes; (void)n_in; (void)out_size;

    k_inproj<<<dim3(MROWS / 64, (2 * D_INNER) / 128), 256, 0, stream>>>(x1, W_in, xz);
    k_conv<<<(MROWS * D_INNER) / 256, 256, 0, stream>>>(xz, conv_w, conv_b, xc);
    k_xproj<<<MROWS / 128, 256, 0, stream>>>(xc, W_x, xdbl);
    k_dt<<<(MROWS * D_INNER) / 256, 256, 0, stream>>>(xdbl, W_dt, b_dt, dtb);
    k_scan<<<64, 128, 0, stream>>>(xz, xc, xdbl, dtb, A_log, Dv, yg);
    k_out_ln<<<MROWS / 16, 256, 0, stream>>>(yg, W_out, ln_g, ln_b, out);
}
